// DyGraphTransformer_87342454931888
// MI455X (gfx1250) — matrix operand order repaired and checked by reading, not compiled, not run on hardware
//
#include <hip/hip_runtime.h>
#include <hip/hip_bf16.h>
#include <math.h>

#define NT_N 20000
#define NC_N 80000
#define NTOT (NT_N + NC_N)
#define HID 128
#define NH 8
#define DK 16
#define EDGES 500000
#define SCALE 0.25f
#define LDA 136   // LDS row stride (bf16 elems): 272B = 16B-aligned, 4-bank skew

typedef __attribute__((ext_vector_type(16))) __bf16 v16bf;
typedef __attribute__((ext_vector_type(8)))  float  v8f;

// ---------------------------------------------------------------------------
// LayerNorm: one wave (32 lanes) per row, 4 floats/lane, wave32 shuffle reduce
// ---------------------------------------------------------------------------
__global__ __launch_bounds__(256)
void ln128_kernel(const float* __restrict__ X, const float* __restrict__ g,
                  const float* __restrict__ b, float* __restrict__ Y, int M)
{
    int wave = threadIdx.x >> 5;
    int lane = threadIdx.x & 31;
    int row  = blockIdx.x * 8 + wave;
    if (row >= M) return;
    const float4* xr = (const float4*)(X + (size_t)row * HID);
    float4 v = xr[lane];
    float s  = v.x + v.y + v.z + v.w;
    float s2 = v.x*v.x + v.y*v.y + v.z*v.z + v.w*v.w;
    #pragma unroll
    for (int m = 16; m > 0; m >>= 1) {
        s  += __shfl_xor(s,  m, 32);
        s2 += __shfl_xor(s2, m, 32);
    }
    float mean = s * (1.0f / HID);
    float var  = s2 * (1.0f / HID) - mean * mean;
    float inv  = rsqrtf(var + 1e-5f);
    float4 gg = ((const float4*)g)[lane];
    float4 bb = ((const float4*)b)[lane];
    float4 o;
    o.x = (v.x - mean) * inv * gg.x + bb.x;
    o.y = (v.y - mean) * inv * gg.y + bb.y;
    o.z = (v.z - mean) * inv * gg.z + bb.z;
    o.w = (v.w - mean) * inv * gg.w + bb.w;
    ((float4*)(Y + (size_t)row * HID))[lane] = o;
}

// ---------------------------------------------------------------------------
// Y = act(X[Mx128] @ W[128x128] + bias) (+ R)   via v_wmma_f32_16x16x32_bf16
// 256 threads = 8 waves; block does 64 rows x 128 cols.
// Wave w owns column tile [16w,16w+16) and 4 row-tiles; B fragment loaded once
// per K-step and reused across the 4 WMMAs.  W staged TRANSPOSED in LDS so
// both A and B fragments are contiguous 16B runs -> ds_load_b128.
// ---------------------------------------------------------------------------
__device__ __forceinline__ float gelu_exact(float x) {
    return 0.5f * x * (1.0f + erff(x * 0.70710678118654752f));
}

__global__ __launch_bounds__(256)
void gemm128_wmma_kernel(const float* __restrict__ X, const float* __restrict__ W,
                         const float* __restrict__ bias, const float* __restrict__ R,
                         float* __restrict__ Y, int M, int act)
{
    __shared__ __bf16 sA [64  * LDA];   // activations, row-major, padded
    __shared__ __bf16 sBt[HID * LDA];   // weights TRANSPOSED: sBt[n*LDA + k]

    const int tid  = threadIdx.x;
    const int row0 = blockIdx.x * 64;

    // stage W transposed (coalesced global read, skewed LDS write)
    for (int i = tid; i < HID * HID; i += 256) {
        int k = i >> 7, n = i & 127;
        sBt[n * LDA + k] = (__bf16)W[i];
    }
    // stage 64 activation rows
    for (int i = tid; i < 64 * HID; i += 256) {
        int r = i >> 7, c = i & 127;
        int gr = row0 + r;
        sA[r * LDA + c] = (__bf16)((gr < M) ? X[(size_t)gr * HID + c] : 0.0f);
    }
    __syncthreads();

    const int wave = tid >> 5;          // 0..7 -> column tile
    const int lane = tid & 31;
    const int g    = lane >> 4;         // lane half-group
    const int ln   = lane & 15;
    const int col0 = wave * 16;

    v8f acc[4] = {{}, {}, {}, {}};
    #pragma unroll
    for (int kk = 0; kk < HID; kk += 32) {
        // B fragment (32x16): lane ln = col N, element e -> K = kk + 16g + e
        // contiguous 16 bf16 in sBt -> 2x ds_load_b128
        v16bf bfr;
        #pragma unroll
        for (int e = 0; e < 16; ++e)
            bfr[e] = sBt[(col0 + ln) * LDA + kk + 16 * g + e];

        #pragma unroll
        for (int rt = 0; rt < 4; ++rt) {
            // A fragment (16x32): lane ln = row M, element e -> K = kk + 16g + e
            // two contiguous 8-elem runs -> 2x ds_load_b128
            v16bf a;
            #pragma unroll
            for (int e = 0; e < 16; ++e) {
                int K = kk + 16 * g + e;
                a[e] = sA[(rt * 16 + ln) * LDA + K];
            }
            acc[rt] = __builtin_amdgcn_wmma_f32_16x16x32_bf16(
                false, a, false, bfr, (short)0, acc[rt], false, false);
        }
    }

    const float bcol = bias[col0 + ln];
    #pragma unroll
    for (int rt = 0; rt < 4; ++rt) {
        #pragma unroll
        for (int r = 0; r < 8; ++r) {
            int row = row0 + rt * 16 + r + 8 * g;   // C/D: lanes16-31 -> M = r+8
            if (row < M) {
                float v = acc[rt][r] + bcol;
                if (act == 1) v = gelu_exact(v);
                size_t off = (size_t)row * HID + col0 + ln;
                if (R) v += R[off];
                Y[off] = v;
            }
        }
    }
}

// ---------------------------------------------------------------------------
// Edge softmax passes (one thread per edge*head)
// ---------------------------------------------------------------------------
__device__ __forceinline__ unsigned enc_f32(float f) {
    unsigned b = __float_as_uint(f);
    return (b & 0x80000000u) ? ~b : (b | 0x80000000u);
}
__device__ __forceinline__ float dec_f32(unsigned u) {
    return (u & 0x80000000u) ? __uint_as_float(u & 0x7FFFFFFFu)
                             : __uint_as_float(~u);
}

__global__ __launch_bounds__(256)
void edge_score_kernel(const float* __restrict__ Q, const float* __restrict__ K,
                       const int* __restrict__ row, const int* __restrict__ col,
                       const float* __restrict__ bias, float* __restrict__ sbuf,
                       unsigned* __restrict__ menc, int qbase, int kbase)
{
    int idx = blockIdx.x * blockDim.x + threadIdx.x;
    if (idx >= EDGES * NH) return;
    int e = idx >> 3, h = idx & 7;
    int r = row[e];
    const float4* qp = (const float4*)(Q + ((size_t)(qbase + r)      * HID + h * DK));
    const float4* kp = (const float4*)(K + ((size_t)(kbase + col[e]) * HID + h * DK));
    float s = 0.0f;
    #pragma unroll
    for (int j = 0; j < 4; ++j) {
        float4 a = qp[j], b = kp[j];
        s += a.x * b.x + a.y * b.y + a.z * b.z + a.w * b.w;
    }
    s = s * SCALE + bias[idx];
    sbuf[idx] = s;
    atomicMax(&menc[(size_t)r * NH + h], enc_f32(s));
}

__global__ __launch_bounds__(256)
void edge_exp_kernel(const int* __restrict__ row, float* __restrict__ sbuf,
                     const unsigned* __restrict__ menc, float* __restrict__ z)
{
    int idx = blockIdx.x * blockDim.x + threadIdx.x;
    if (idx >= EDGES * NH) return;
    int e = idx >> 3, h = idx & 7;
    float m  = dec_f32(menc[(size_t)row[e] * NH + h]);
    float ex = expf(sbuf[idx] - m);
    sbuf[idx] = ex;
    atomicAdd(&z[(size_t)row[e] * NH + h], ex);
}

__global__ __launch_bounds__(256)
void edge_accum_kernel(const int* __restrict__ row, const int* __restrict__ col,
                       const float* __restrict__ sbuf, const float* __restrict__ z,
                       const float* __restrict__ V, float* __restrict__ obuf,
                       int vbase, int obase)
{
    int idx = blockIdx.x * blockDim.x + threadIdx.x;
    if (idx >= EDGES * NH) return;
    int e = idx >> 3, h = idx & 7;
    float a = sbuf[idx] / (z[(size_t)row[e] * NH + h] + 1e-16f);
    const float* vp = V    + ((size_t)(vbase + col[e]) * HID + h * DK);
    float*       op = obuf + ((size_t)(obase + row[e]) * HID + h * DK);
    #pragma unroll
    for (int j = 0; j < DK; ++j)
        atomicAdd(&op[j], a * vp[j]);
}

// ---------------------------------------------------------------------------
extern "C" void kernel_launch(void* const* d_in, const int* in_sizes, int n_in,
                              void* d_out, int out_size, void* d_ws, size_t ws_size,
                              hipStream_t stream) {
    const float* x_all    = (const float*)d_in[0];
    const float* bias_c2t = (const float*)d_in[1];
    const float* bias_t2c = (const float*)d_in[2];
    const int*   c2t_row  = (const int*)d_in[3];
    const int*   c2t_col  = (const int*)d_in[4];
    const int*   t2c_row  = (const int*)d_in[5];
    const int*   t2c_col  = (const int*)d_in[6];
    const float* ln1_g = (const float*)d_in[7];
    const float* ln1_b = (const float*)d_in[8];
    const float* Wq = (const float*)d_in[9];  const float* bq = (const float*)d_in[10];
    const float* Wk = (const float*)d_in[11]; const float* bk = (const float*)d_in[12];
    const float* Wv = (const float*)d_in[13]; const float* bv = (const float*)d_in[14];
    const float* Wo = (const float*)d_in[15]; const float* bo = (const float*)d_in[16];
    const float* ln2_g = (const float*)d_in[17];
    const float* ln2_b = (const float*)d_in[18];
    const float* W1 = (const float*)d_in[19]; const float* b1 = (const float*)d_in[20];
    const float* W2 = (const float*)d_in[21]; const float* b2 = (const float*)d_in[22];
    float* out = (float*)d_out;

    const size_t NF = (size_t)NTOT * HID;
    float* ws   = (float*)d_ws;
    float* y    = ws;            // LN1 out; later LN2 out
    float* Qb   = ws + 1 * NF;   // later: gelu hidden
    float* Kb   = ws + 2 * NF;
    float* Vb   = ws + 3 * NF;
    float* obuf = ws + 4 * NF;   // attn accumulators in y_all concat order
    float* xbuf = ws + 5 * NF;   // x = x_all + y_all
    float* sb0  = ws + 6 * NF;                        // E*NH scores dir0
    float* sb1  = sb0 + (size_t)EDGES * NH;           // E*NH scores dir1
    unsigned* m0 = (unsigned*)(sb1 + (size_t)EDGES * NH); // NT*NH
    unsigned* m1 = m0 + (size_t)NT_N * NH;                // NC*NH
    float* z0 = (float*)(m1 + (size_t)NC_N * NH);         // NT*NH
    float* z1 = z0 + (size_t)NT_N * NH;                   // NC*NH

    const dim3 blk(256);
    const dim3 gLN((NTOT + 7) / 8);
    const dim3 gGEMM((NTOT + 63) / 64);
    const dim3 gEdge((EDGES * NH + 255) / 256);

    // 1. y = LN1(x_all)
    ln128_kernel<<<gLN, blk, 0, stream>>>(x_all, ln1_g, ln1_b, y, NTOT);

    // 2. Q/K/V for all nodes (covers both attention directions)
    gemm128_wmma_kernel<<<gGEMM, blk, 0, stream>>>(y, Wq, bq, nullptr, Qb, NTOT, 0);
    gemm128_wmma_kernel<<<gGEMM, blk, 0, stream>>>(y, Wk, bk, nullptr, Kb, NTOT, 0);
    gemm128_wmma_kernel<<<gGEMM, blk, 0, stream>>>(y, Wv, bv, nullptr, Vb, NTOT, 0);

    // 3. zero segment-max encodings, Z sums, attention accumulators
    hipMemsetAsync(m0,   0, sizeof(unsigned) * (size_t)NTOT * NH, stream);
    hipMemsetAsync(z0,   0, sizeof(float)    * (size_t)NTOT * NH, stream);
    hipMemsetAsync(obuf, 0, sizeof(float)    * NF,                stream);

    // 4. scores + segment max.
    // dir0 (cxt2tgt): q rows = targets [0,NT), k/v rows = contexts [NT,NTOT)
    edge_score_kernel<<<gEdge, blk, 0, stream>>>(Qb, Kb, c2t_row, c2t_col,
                                                 bias_c2t, sb0, m0, 0, NT_N);
    // dir1 (tgt2cxt): q rows = contexts, k/v rows = targets
    edge_score_kernel<<<gEdge, blk, 0, stream>>>(Qb, Kb, t2c_row, t2c_col,
                                                 bias_t2c, sb1, m1, NT_N, 0);

    // 5. exp + segment sum
    edge_exp_kernel<<<gEdge, blk, 0, stream>>>(c2t_row, sb0, m0, z0);
    edge_exp_kernel<<<gEdge, blk, 0, stream>>>(t2c_row, sb1, m1, z1);

    // 6. weighted-V scatter. y_all = concat([attn2 out (NC rows), attn1 out (NT rows)])
    edge_accum_kernel<<<gEdge, blk, 0, stream>>>(c2t_row, c2t_col, sb0, z0, Vb,
                                                 obuf, NT_N, NC_N); // attn1 -> rows [NC,NTOT)
    edge_accum_kernel<<<gEdge, blk, 0, stream>>>(t2c_row, t2c_col, sb1, z1, Vb,
                                                 obuf, 0, 0);       // attn2 -> rows [0,NC)

    // 7. x = x_all + obuf @ Wo + bo
    gemm128_wmma_kernel<<<gGEMM, blk, 0, stream>>>(obuf, Wo, bo, x_all, xbuf, NTOT, 0);

    // 8. LN2
    ln128_kernel<<<gLN, blk, 0, stream>>>(xbuf, ln2_g, ln2_b, y, NTOT);

    // 9. hidden = gelu(y @ W1 + b1)
    gemm128_wmma_kernel<<<gGEMM, blk, 0, stream>>>(y, W1, b1, nullptr, Qb, NTOT, 1);

    // 10. out = x + hidden @ W2 + b2
    gemm128_wmma_kernel<<<gGEMM, blk, 0, stream>>>(Qb, W2, b2, xbuf, out, NTOT, 0);
}
